// DynamicConv_64905545777402
// MI455X (gfx1250) — matrix operand order repaired and checked by reading, not compiled, not run on hardware
//
#include <hip/hip_runtime.h>

typedef __attribute__((ext_vector_type(16))) _Float16 v16h;
typedef __attribute__((ext_vector_type(8)))  _Float16 v8h;
typedef __attribute__((ext_vector_type(8)))  float    v8f;

#define B_    32
#define C_    64
#define CP    72                  // channel-minor LDS pitch (pad 64 -> 72)
#define H_    128
#define W_    128
#define OUT_  64
#define HID   128
#define KTOT  576                 // C_ * 3 * 3
#define NW    (OUT_ * KTOT)       // 36864 generated weights per sample
#define ROWS  2                   // output rows per workgroup

// ---------------------------------------------------------------------------
// Kernel 1: per-(b,c) spatial mean over 128x128
// ---------------------------------------------------------------------------
__global__ void __launch_bounds__(256) k_stats(const float* __restrict__ x,
                                               float* __restrict__ stats) {
  const int c = blockIdx.x, b = blockIdx.y;
  const float* p = x + ((size_t)(b * C_ + c)) * (H_ * W_);
  float s = 0.f;
  for (int i = threadIdx.x; i < H_ * W_; i += 256) s += p[i];
  __shared__ float red[256];
  red[threadIdx.x] = s;
  __syncthreads();
  for (int st = 128; st > 0; st >>= 1) {
    if (threadIdx.x < st) red[threadIdx.x] += red[threadIdx.x + st];
    __syncthreads();
  }
  if (threadIdx.x == 0) stats[b * C_ + c] = red[0] * (1.0f / (H_ * W_));
}

// ---------------------------------------------------------------------------
// Kernel 2: hidden = relu(stats @ W1 + b1)   [32 x 128]  (tiny; VALU is fine)
// ---------------------------------------------------------------------------
__global__ void __launch_bounds__(256) k_mlp1(const float* __restrict__ stats,
                                              const float* __restrict__ W1,
                                              const float* __restrict__ b1,
                                              _Float16* __restrict__ hidden_h) {
  const int gid = blockIdx.x * 256 + threadIdx.x;   // 0..4095
  const int i = gid >> 7;                           // batch 0..31
  const int j = gid & 127;                          // hidden unit
  float s = b1[j];
  for (int c = 0; c < C_; ++c) s += stats[i * C_ + c] * W1[c * HID + j];
  s = fmaxf(s, 0.f);
  hidden_h[i * HID + j] = (_Float16)s;
}

// ---------------------------------------------------------------------------
// Kernel 3: weight = hidden @ W2 + b2  via WMMA f16 (M=32, K=128, N=36864).
// Output stored f16 per sample in TAP-MAJOR layout w[tap][out][c] so the conv
// kernel's A tiles are contiguous 16B runs in LDS.
// ---------------------------------------------------------------------------
__global__ void __launch_bounds__(256) k_wgen(const _Float16* __restrict__ hidden_h,
                                              const float* __restrict__ W2,
                                              const float* __restrict__ b2,
                                              _Float16* __restrict__ wf16) {
  const int lane = threadIdx.x & 31;
  const int wave = threadIdx.x >> 5;                 // 8 waves -> 8 n-strips
  const int n = blockIdx.x * 128 + wave * 16 + (lane & 15);
  const int klo = (lane >= 16) ? 16 : 0;             // unused; B gather uses ka too
  const int ka  = (lane >= 16) ? 8  : 0;             // A-matrix K base per lane half
  const int m0  = lane & 15;

  v8f acc0 = {}, acc1 = {};
  for (int ks = 0; ks < 4; ++ks) {                   // K = 128 in 4 steps of 32
    // B tile [32 x 16] gathered from W2 (f32 -> f16); lanes 0-15 contiguous in n
    v16h bt;
#pragma unroll
    for (int h = 0; h < 16; ++h) {
      const int k = ks * 32 + ka + (h < 8 ? h : 8 + h);
      bt[h] = (_Float16)W2[(size_t)k * NW + n];
    }
    // A tiles [16 x 32] from hidden_h: two contiguous 8-half runs per lane
    const int kb = ks * 32 + ka;
    v16h a0, a1;
    {
      v8h lo = *(const v8h*)(hidden_h + m0 * HID + kb);
      v8h hi = *(const v8h*)(hidden_h + m0 * HID + kb + 16);
#pragma unroll
      for (int t = 0; t < 8; ++t) { a0[t] = lo[t]; a0[t + 8] = hi[t]; }
      v8h lo1 = *(const v8h*)(hidden_h + (16 + m0) * HID + kb);
      v8h hi1 = *(const v8h*)(hidden_h + (16 + m0) * HID + kb + 16);
#pragma unroll
      for (int t = 0; t < 8; ++t) { a1[t] = lo1[t]; a1[t + 8] = hi1[t]; }
    }
    acc0 = __builtin_amdgcn_wmma_f32_16x16x32_f16(false, a0, false, bt, (short)0, acc0, false, false);
    acc1 = __builtin_amdgcn_wmma_f32_16x16x32_f16(false, a1, false, bt, (short)0, acc1, false, false);
  }

  // n -> (o, c, tap); store tap-major: dst = m*NW + (tap*OUT_ + o)*C_ + c
  const int o   = n / KTOT;
  const int r   = n - o * KTOT;
  const int cch = r / 9;
  const int tap = r - cch * 9;
  const size_t dofs = (size_t)(tap * OUT_ + o) * C_ + cch;

  const float bias = b2[n];
  const int mofs = (lane >= 16) ? 8 : 0;
#pragma unroll
  for (int rr = 0; rr < 8; ++rr) {
    const int m = rr + mofs;                         // batch index 0..15 / 16..31
    wf16[(size_t)m        * NW + dofs] = (_Float16)(acc0[rr] + bias);
    wf16[(size_t)(m + 16) * NW + dofs] = (_Float16)(acc1[rr] + bias);
  }
}

// ---------------------------------------------------------------------------
// Kernel 4: per-sample 3x3 conv as implicit GEMM, tap-decomposed:
//   for each tap (ky,kx): D[64 out, 128 px] += Wtap[64,64] * Xtap[64,128]
// One workgroup per (2 output rows, sample). 8 waves: wave -> 16-px n-tile,
// each wave does all 4 M-tiles x 2 rows; 9 taps x 2 c-chunks of K=32.
// All LDS traffic is contiguous ds_load_b128.
// ---------------------------------------------------------------------------
__global__ void __launch_bounds__(256, 1) k_conv(const float* __restrict__ x,
                                                 const _Float16* __restrict__ wf16,
                                                 float* __restrict__ out) {
  const int hb = blockIdx.x * ROWS;                  // first output row
  const int b  = blockIdx.y;                         // sample

  __shared__ __align__(16) _Float16 wlds[9 * OUT_ * C_];        // 72 KB, [tap][out][c]
  __shared__ __align__(16) _Float16 xpadT[(ROWS + 2) * 130 * CP]; // 74.9 KB, [row][col][c]

  // Stage this sample's generated weights (already tap-major f16) into LDS.
  {
    const uint* src = (const uint*)(wf16 + (size_t)b * NW);
    uint* dst = (uint*)wlds;
    for (int i = threadIdx.x; i < NW / 2; i += 256) dst[i] = src[i];
  }
  // Stage ROWS+2 input rows x 64 ch, f32->f16, channel-minor with halo pad.
  // Global reads stay coalesced (col fast-varying per channel row).
  {
    const float* xb = x + ((size_t)b * C_) * (H_ * W_);
    const int NROW = ROWS + 2;
    for (int i = threadIdx.x; i < C_ * NROW * 130; i += 256) {
      const int c   = i / (NROW * 130);
      const int rr  = i - c * (NROW * 130);
      const int ry  = rr / 130;
      const int col = rr - ry * 130;
      const int row = hb + ry - 1;
      float v = 0.f;
      if (col >= 1 && col <= 128 && row >= 0 && row < H_)
        v = xb[((size_t)c * H_ + row) * W_ + (col - 1)];
      xpadT[(ry * 130 + col) * CP + c] = (_Float16)v;
    }
  }
  __syncthreads();

  const int lane = threadIdx.x & 31;
  const int wave = threadIdx.x >> 5;                 // n-tile 0..7
  const int wn   = wave * 16 + (lane & 15);          // this lane's pixel column
  const int klo  = (lane >= 16) ? 16 : 0;            // B-layout K base
  const int ka   = (lane >= 16) ? 8  : 0;            // unused; A tiles use klo too
  const int m0   = lane & 15;

  v8f acc[ROWS][4] = {};

#pragma unroll
  for (int t = 0; t < 9; ++t) {                      // tap loop
    const int ky = t / 3;
    const int kx = t - ky * 3;
#pragma unroll
    for (int cc = 0; cc < 2; ++cc) {                 // K = 64 in chunks of 32
      const int cb = cc * 32;
      // B tiles for both output rows: contiguous 32B run per lane.
      const _Float16* b0 = xpadT + (ky * 130 + wn + kx) * CP + cb + klo;
      v16h bt0, bt1;
      {
        v8h lo = *(const v8h*)b0;
        v8h hi = *(const v8h*)(b0 + 8);
#pragma unroll
        for (int q = 0; q < 8; ++q) { bt0[q] = lo[q]; bt0[q + 8] = hi[q]; }
        const _Float16* b1 = b0 + 130 * CP;          // next input row
        v8h lo1 = *(const v8h*)b1;
        v8h hi1 = *(const v8h*)(b1 + 8);
#pragma unroll
        for (int q = 0; q < 8; ++q) { bt1[q] = lo1[q]; bt1[q + 8] = hi1[q]; }
      }
      // A tiles (weights, tap-major) reused across both rows.
      const _Float16* ab = wlds + (t * OUT_ + m0) * C_ + cb + klo;
#pragma unroll
      for (int mt = 0; mt < 4; ++mt) {
        v8h lo = *(const v8h*)(ab + mt * 16 * C_);
        v8h hi = *(const v8h*)(ab + mt * 16 * C_ + 8);
        v16h at;
#pragma unroll
        for (int q = 0; q < 8; ++q) { at[q] = lo[q]; at[q + 8] = hi[q]; }
        acc[0][mt] = __builtin_amdgcn_wmma_f32_16x16x32_f16(false, at, false, bt0,
                                                            (short)0, acc[0][mt], false, false);
        acc[1][mt] = __builtin_amdgcn_wmma_f32_16x16x32_f16(false, at, false, bt1,
                                                            (short)0, acc[1][mt], false, false);
      }
    }
  }

  // Epilogue: D layout -> out[b][o][hb+row][w]
  const int mofs = (lane >= 16) ? 8 : 0;
#pragma unroll
  for (int row = 0; row < ROWS; ++row) {
#pragma unroll
    for (int mt = 0; mt < 4; ++mt) {
#pragma unroll
      for (int r = 0; r < 8; ++r) {
        const int o = mt * 16 + r + mofs;
        out[(((size_t)b * OUT_ + o) * H_ + (hb + row)) * W_ + wn] = acc[row][mt][r];
      }
    }
  }
}

// ---------------------------------------------------------------------------
extern "C" void kernel_launch(void* const* d_in, const int* in_sizes, int n_in,
                              void* d_out, int out_size, void* d_ws, size_t ws_size,
                              hipStream_t stream) {
  const float* x  = (const float*)d_in[0];
  const float* W1 = (const float*)d_in[1];
  const float* b1 = (const float*)d_in[2];
  const float* W2 = (const float*)d_in[3];
  const float* b2 = (const float*)d_in[4];
  float* out = (float*)d_out;

  char* ws = (char*)d_ws;
  float*    stats    = (float*)ws;                      // 2048 f32   (8 KB)
  _Float16* hidden_h = (_Float16*)(ws + 16 * 1024);     // 4096 f16   (8 KB)
  _Float16* wf16     = (_Float16*)(ws + 32 * 1024);     // 32*36864 f16 (2.36 MB)

  k_stats<<<dim3(C_, B_), 256, 0, stream>>>(x, stats);
  k_mlp1 <<<16, 256, 0, stream>>>(stats, W1, b1, hidden_h);
  k_wgen <<<NW / 128, 256, 0, stream>>>(hidden_h, W2, b2, wf16);
  k_conv <<<dim3(H_ / ROWS, B_), 256, 0, stream>>>(x, wf16, out);
}
